// ChildSumTreeLSTM_13795434954927
// MI455X (gfx1250) — matrix operand order repaired and checked by reading, not compiled, not run on hardware
//
#include <hip/hip_runtime.h>
#include <hip/hip_bf16.h>

// ---------------------------------------------------------------------------
// Child-Sum TreeLSTM on a perfect binary tree, level-parallel, bf16 WMMA.
// N = 8191 = 2^13 - 1, M = D = 256.
// topo index t: leaves t in [0,4096); level d (root=0) occupies
//   t in [8192 - 2^(d+1), 8191 - 2^d], children of t: 2t-8191, 2t-8192.
// GEMMs: one wave computes a 16x64 output strip (4 accumulators), reusing the
// A fragment across 4 B tiles -> 32 WMMAs per wave, 4x less A traffic.
// ---------------------------------------------------------------------------

#define N_NODES 8191

typedef __attribute__((ext_vector_type(16))) __bf16 v16bf;
typedef __attribute__((ext_vector_type(8)))  float  v8f;

__device__ __forceinline__ float sigmoidf(float x) {
  return 1.0f / (1.0f + expf(-x));
}

// A fragment: 16x32 bf16 tile (M=rows, K=cols). Per-lane row pointer (gather-
// friendly). lanes 0-15: M=lane, K = {k0..k0+7} U {k0+16..k0+23}
//            lanes 16-31: M=lane-16, K = {k0+8..k0+15} U {k0+24..k0+31}
__device__ __forceinline__ v16bf load_a_frag(const __bf16* rowbase, int k0, int lane) {
  union { v16bf v; float4 f4[2]; } u;
  int hi = lane >> 4;
  u.f4[0] = *(const float4*)(rowbase + k0 + hi * 8);
  u.f4[1] = *(const float4*)(rowbase + k0 + 16 + hi * 8);
  return u.v;
}

// B fragment: 32x16 bf16 tile (K=rows, N=cols), B(K,N) = W[n0+N, k0+K]
// (W row-major [out_feat, in_feat]). lanes 0-15: N=lane, K = {k0..k0+7} U {k0+16..k0+23};
// lanes 16-31: N=lane-16, K = {k0+8..k0+15} U {k0+24..k0+31}. Two 16-byte loads.
__device__ __forceinline__ v16bf load_b_frag(const __bf16* wrowbase, int k0, int lane) {
  union { v16bf v; float4 f4[2]; } u;
  int hi = lane >> 4;
  const __bf16* p = wrowbase + k0 + hi * 8;
  u.f4[0] = *(const float4*)(p);
  u.f4[1] = *(const float4*)(p + 16);
  return u.v;
}

// ---------------------------------------------------------------------------
// Prep kernels
// ---------------------------------------------------------------------------

// Wcat[1024,256] = concat(W_ioux[768,256], W_fx[256,256]) in bf16;
// Wiouh_bf, Wfh_bf straight conversions.
__global__ void convert_weights(const float* __restrict__ W_ioux,
                                const float* __restrict__ W_iouh,
                                const float* __restrict__ W_fx,
                                const float* __restrict__ W_fh,
                                __bf16* __restrict__ Wcat,
                                __bf16* __restrict__ Wiouh_bf,
                                __bf16* __restrict__ Wfh_bf) {
  int idx = blockIdx.x * blockDim.x + threadIdx.x;  // 2048*256 total
  const int A = 1024 * 256, B = A + 768 * 256, C = B + 256 * 256;
  if (idx < A) {
    float v = (idx < 768 * 256) ? W_ioux[idx] : W_fx[idx - 768 * 256];
    Wcat[idx] = (__bf16)v;
  } else if (idx < B) {
    int j = idx - A; Wiouh_bf[j] = (__bf16)W_iouh[j];
  } else if (idx < C) {
    int j = idx - B; Wfh_bf[j] = (__bf16)W_fh[j];
  }
}

// iou_leaf[m] = ones @ W_iouh.T + b_iouh (m<768);  leaf_fh[m] = ones @ W_fh.T + b_fh
__global__ void leaf_vecs(const float* __restrict__ W_iouh, const float* __restrict__ b_iouh,
                          const float* __restrict__ W_fh,   const float* __restrict__ b_fh,
                          float* __restrict__ iou_leaf, float* __restrict__ leaf_fh) {
  int m = blockIdx.x * blockDim.x + threadIdx.x;  // 1024
  if (m < 768) {
    float s = b_iouh[m];
    for (int d = 0; d < 256; ++d) s += W_iouh[m * 256 + d];
    iou_leaf[m] = s;
  } else if (m < 1024) {
    int j = m - 768;
    float s = b_fh[j];
    for (int d = 0; d < 256; ++d) s += W_fh[j * 256 + d];
    leaf_fh[j] = s;
  }
}

// Xg[t,d] = bf16(emb[ops[t], d])
__global__ void gather_embed(const float* __restrict__ emb, const int* __restrict__ ops,
                             __bf16* __restrict__ Xg) {
  int idx = blockIdx.x * blockDim.x + threadIdx.x;  // N*256
  if (idx >= N_NODES * 256) return;
  int t = idx >> 8, d = idx & 255;
  Xg[idx] = (__bf16)emb[(size_t)ops[t] * 256 + d];
}

// ---------------------------------------------------------------------------
// Precompute GEMM: ioufx[N,1024] = Xg[N,256] @ Wcat[1024,256]^T + bias
// one wave per 16x64 strip; grid = (16 col groups, 512 row tiles)
// ---------------------------------------------------------------------------
__global__ void gemm_pre(const __bf16* __restrict__ Xg, const __bf16* __restrict__ Wcat,
                         const float* __restrict__ b_ioux, const float* __restrict__ b_fx,
                         float* __restrict__ ioufx) {
  int lane = threadIdx.x;           // blockDim = 32, EXEC all ones
  int g  = blockIdx.x;              // 0..15 : 64-column group
  int rt = blockIdx.y;              // 0..511
  int m = lane & 15, hi = lane >> 4;

  int node = rt * 16 + m; if (node >= N_NODES) node = N_NODES - 1;  // pad rows discarded
  const __bf16* arow = Xg + (size_t)node * 256;
  const __bf16* brow0 = Wcat + (size_t)((g * 4 + 0) * 16 + m) * 256;
  const __bf16* brow1 = Wcat + (size_t)((g * 4 + 1) * 16 + m) * 256;
  const __bf16* brow2 = Wcat + (size_t)((g * 4 + 2) * 16 + m) * 256;
  const __bf16* brow3 = Wcat + (size_t)((g * 4 + 3) * 16 + m) * 256;

  v8f acc0 = {}, acc1 = {}, acc2 = {}, acc3 = {};
#pragma unroll
  for (int k0 = 0; k0 < 256; k0 += 32) {
    v16bf a = load_a_frag(arow, k0, lane);
    acc0 = __builtin_amdgcn_wmma_f32_16x16x32_bf16(false, a, false, load_b_frag(brow0, k0, lane), (short)0, acc0, false, false);
    acc1 = __builtin_amdgcn_wmma_f32_16x16x32_bf16(false, a, false, load_b_frag(brow1, k0, lane), (short)0, acc1, false, false);
    acc2 = __builtin_amdgcn_wmma_f32_16x16x32_bf16(false, a, false, load_b_frag(brow2, k0, lane), (short)0, acc2, false, false);
    acc3 = __builtin_amdgcn_wmma_f32_16x16x32_bf16(false, a, false, load_b_frag(brow3, k0, lane), (short)0, acc3, false, false);
  }

  v8f accs[4] = {acc0, acc1, acc2, acc3};
#pragma unroll
  for (int j = 0; j < 4; ++j) {
    int col = (g * 4 + j) * 16 + m;
    float bias = (col < 768) ? b_ioux[col] : b_fx[col - 768];
#pragma unroll
    for (int v = 0; v < 8; ++v) {
      int r = rt * 16 + v + 8 * hi;
      if (r < N_NODES) ioufx[(size_t)r * 1024 + col] = accs[j][v] + bias;
    }
  }
}

// ---------------------------------------------------------------------------
// Leaves: iou = ioufx[t,0:768] + iou_leaf;  fc = sigmoid(leaf_fh + fx[t])
// ---------------------------------------------------------------------------
__global__ void leaf_nodes(const float* __restrict__ ioufx,
                           const float* __restrict__ iou_leaf, const float* __restrict__ leaf_fh,
                           float* __restrict__ c_arr, float* __restrict__ h_f32,
                           __bf16* __restrict__ h_bf) {
  int idx = blockIdx.x * blockDim.x + threadIdx.x;  // 4096*256
  if (idx >= 4096 * 256) return;
  int t = idx >> 8, d = idx & 255;
  const float* io = ioufx + (size_t)t * 1024;
  float iv = sigmoidf(io[d]       + iou_leaf[d]);
  float ov = sigmoidf(io[256 + d] + iou_leaf[256 + d]);
  float uv = tanhf   (io[512 + d] + iou_leaf[512 + d]);
  float fc = sigmoidf(leaf_fh[d] + io[768 + d]);
  float c = iv * uv + fc;
  float h = ov * tanhf(c);
  size_t o = (size_t)t * 256 + d;
  c_arr[o] = c; h_f32[o] = h; h_bf[o] = (__bf16)h;
}

// hsum_bf[t] = bf16(h[c0] + h[c1]) for t in level
__global__ void hsum_level(const float* __restrict__ h_f32, __bf16* __restrict__ hsum_bf,
                           int t0, int L) {
  int idx = blockIdx.x * blockDim.x + threadIdx.x;
  if (idx >= L * 256) return;
  int t = t0 + (idx >> 8), d = idx & 255;
  int c0 = 2 * t - 8191, c1 = 2 * t - 8192;
  hsum_bf[(size_t)t * 256 + d] =
      (__bf16)(h_f32[(size_t)c0 * 256 + d] + h_f32[(size_t)c1 * 256 + d]);
}

// ---------------------------------------------------------------------------
// Level GEMM: gbuf[r, 0:1280] for r = t - t0:
//   cols [0,768)     : h_sum[t]    @ W_iouh^T   (col groups 0..11)
//   cols [768,1024)  : h[child0]   @ W_fh^T     (col groups 12..15)
//   cols [1024,1280) : h[child1]   @ W_fh^T     (col groups 16..19)
// grid = (20 col groups of 64, ceil(L/16) row tiles), one wave per 16x64 strip
// ---------------------------------------------------------------------------
__global__ void gemm_level(const __bf16* __restrict__ hsum_bf, const __bf16* __restrict__ h_bf,
                           const __bf16* __restrict__ Wiouh_bf, const __bf16* __restrict__ Wfh_bf,
                           float* __restrict__ gbuf, int t0, int L) {
  int lane = threadIdx.x;           // blockDim = 32
  int g  = blockIdx.x;              // 0..19
  int rt = blockIdx.y;
  int m = lane & 15, hi = lane >> 4;

  int r = rt * 16 + m; if (r >= L) r = L - 1;   // pad rows discarded
  int t = t0 + r;

  const __bf16* arow;
  const __bf16* wbase;
  int coff;
  if (g < 12) {                                   // uniform per wave
    arow = hsum_bf + (size_t)t * 256;  wbase = Wiouh_bf; coff = 0;
  } else if (g < 16) {
    arow = h_bf + (size_t)(2 * t - 8191) * 256;   // child0
    wbase = Wfh_bf; coff = 768;
  } else {
    arow = h_bf + (size_t)(2 * t - 8192) * 256;   // child1
    wbase = Wfh_bf; coff = 1024;
  }
  const __bf16* brow0 = wbase + (size_t)(g * 64 + 0 * 16 + m - coff) * 256;
  const __bf16* brow1 = wbase + (size_t)(g * 64 + 1 * 16 + m - coff) * 256;
  const __bf16* brow2 = wbase + (size_t)(g * 64 + 2 * 16 + m - coff) * 256;
  const __bf16* brow3 = wbase + (size_t)(g * 64 + 3 * 16 + m - coff) * 256;

  v8f acc0 = {}, acc1 = {}, acc2 = {}, acc3 = {};
#pragma unroll
  for (int k0 = 0; k0 < 256; k0 += 32) {
    v16bf a = load_a_frag(arow, k0, lane);
    acc0 = __builtin_amdgcn_wmma_f32_16x16x32_bf16(false, a, false, load_b_frag(brow0, k0, lane), (short)0, acc0, false, false);
    acc1 = __builtin_amdgcn_wmma_f32_16x16x32_bf16(false, a, false, load_b_frag(brow1, k0, lane), (short)0, acc1, false, false);
    acc2 = __builtin_amdgcn_wmma_f32_16x16x32_bf16(false, a, false, load_b_frag(brow2, k0, lane), (short)0, acc2, false, false);
    acc3 = __builtin_amdgcn_wmma_f32_16x16x32_bf16(false, a, false, load_b_frag(brow3, k0, lane), (short)0, acc3, false, false);
  }

  v8f accs[4] = {acc0, acc1, acc2, acc3};
#pragma unroll
  for (int j = 0; j < 4; ++j) {
    int col = g * 64 + j * 16 + m;
#pragma unroll
    for (int v = 0; v < 8; ++v) {
      int rr = rt * 16 + v + 8 * hi;
      if (rr < L) gbuf[(size_t)rr * 1280 + col] = accs[j][v];
    }
  }
}

// ---------------------------------------------------------------------------
// Level combine: gates + cell/hidden update for internal nodes
// ---------------------------------------------------------------------------
__global__ void combine_level(const float* __restrict__ gbuf, const float* __restrict__ ioufx,
                              const float* __restrict__ b_iouh, const float* __restrict__ b_fh,
                              float* __restrict__ c_arr, float* __restrict__ h_f32,
                              __bf16* __restrict__ h_bf, int t0, int L) {
  int idx = blockIdx.x * blockDim.x + threadIdx.x;
  if (idx >= L * 256) return;
  int r = idx >> 8, d = idx & 255;
  int t = t0 + r;
  int c0 = 2 * t - 8191, c1 = 2 * t - 8192;
  const float* g  = gbuf  + (size_t)r * 1280;
  const float* io = ioufx + (size_t)t * 1024;
  float iv = sigmoidf(io[d]       + g[d]       + b_iouh[d]);
  float ov = sigmoidf(io[256 + d] + g[256 + d] + b_iouh[256 + d]);
  float uv = tanhf   (io[512 + d] + g[512 + d] + b_iouh[512 + d]);
  float fxv = io[768 + d];
  float f0 = sigmoidf(g[768 + d]  + b_fh[d] + fxv);
  float f1 = sigmoidf(g[1024 + d] + b_fh[d] + fxv);
  float c = iv * uv + f0 * c_arr[(size_t)c0 * 256 + d] + f1 * c_arr[(size_t)c1 * 256 + d];
  float h = ov * tanhf(c);
  size_t o = (size_t)t * 256 + d;
  c_arr[o] = c; h_f32[o] = h; h_bf[o] = (__bf16)h;
}

__global__ void write_out(const float* __restrict__ c_arr, const float* __restrict__ h_f32,
                          float* __restrict__ out) {
  int m = blockIdx.x * blockDim.x + threadIdx.x;  // 512
  const size_t root = (size_t)(N_NODES - 1) * 256;
  if (m < 256)      out[m] = c_arr[root + m];
  else if (m < 512) out[m] = h_f32[root + (m - 256)];
}

// ---------------------------------------------------------------------------
extern "C" void kernel_launch(void* const* d_in, const int* in_sizes, int n_in,
                              void* d_out, int out_size, void* d_ws, size_t ws_size,
                              hipStream_t stream) {
  (void)in_sizes; (void)n_in; (void)out_size; (void)ws_size;
  const float* emb     = (const float*)d_in[0];
  const float* W_ioux  = (const float*)d_in[1];
  const float* b_ioux  = (const float*)d_in[2];
  const float* W_iouh  = (const float*)d_in[3];
  const float* b_iouh  = (const float*)d_in[4];
  const float* W_fx    = (const float*)d_in[5];
  const float* b_fx    = (const float*)d_in[6];
  const float* W_fh    = (const float*)d_in[7];
  const float* b_fh    = (const float*)d_in[8];
  const int*   ops     = (const int*)d_in[9];
  // d_in[10] (children) / d_in[11] (child_mask): perfect heap => derived analytically
  float* out = (float*)d_out;

  char* ws = (char*)d_ws;
  size_t off = 0;
  auto alloc = [&](size_t bytes) -> char* {
    char* p = ws + off; off += (bytes + 255) & ~(size_t)255; return p;
  };
  const size_t N = N_NODES;
  float*  ioufx    = (float*) alloc(N * 1024 * 4);   // [:,0:768]=iou_x, [:,768:]=fx
  float*  c_arr    = (float*) alloc(N * 256 * 4);
  float*  h_f32    = (float*) alloc(N * 256 * 4);
  __bf16* h_bf     = (__bf16*)alloc(N * 256 * 2);
  __bf16* hsum_bf  = (__bf16*)alloc(N * 256 * 2);
  __bf16* Xg       = (__bf16*)alloc(N * 256 * 2);
  __bf16* Wcat     = (__bf16*)alloc(1024 * 256 * 2);
  __bf16* Wiouh_bf = (__bf16*)alloc(768 * 256 * 2);
  __bf16* Wfh_bf   = (__bf16*)alloc(256 * 256 * 2);
  float*  iou_leaf = (float*) alloc(768 * 4);
  float*  leaf_fh  = (float*) alloc(256 * 4);
  float*  gbuf     = (float*) alloc(2048 * 1280 * 4);

  // 1) weight conversion + leaf vectors + embedding gather
  convert_weights<<<(2048 * 256) / 256, 256, 0, stream>>>(W_ioux, W_iouh, W_fx, W_fh,
                                                          Wcat, Wiouh_bf, Wfh_bf);
  leaf_vecs<<<4, 256, 0, stream>>>(W_iouh, b_iouh, W_fh, b_fh, iou_leaf, leaf_fh);
  gather_embed<<<(N_NODES * 256 + 255) / 256, 256, 0, stream>>>(emb, ops, Xg);

  // 2) precompute GEMM: [8191,256] x [256,1024] via bf16 WMMA (16x64 per wave)
  gemm_pre<<<dim3(16, 512), 32, 0, stream>>>(Xg, Wcat, b_ioux, b_fx, ioufx);

  // 3) leaves (t = 0..4095): pure elementwise
  leaf_nodes<<<(4096 * 256) / 256, 256, 0, stream>>>(ioufx, iou_leaf, leaf_fh,
                                                     c_arr, h_f32, h_bf);

  // 4) internal levels, bottom-up (d = 11 .. 0)
  for (int d = 11; d >= 0; --d) {
    int L  = 1 << d;
    int t0 = 8192 - (1 << (d + 1));
    int elems = L * 256;
    hsum_level<<<(elems + 255) / 256, 256, 0, stream>>>(h_f32, hsum_bf, t0, L);
    gemm_level<<<dim3(20, (L + 15) / 16), 32, 0, stream>>>(hsum_bf, h_bf, Wiouh_bf, Wfh_bf,
                                                           gbuf, t0, L);
    combine_level<<<(elems + 255) / 256, 256, 0, stream>>>(gbuf, ioufx, b_iouh, b_fh,
                                                           c_arr, h_f32, h_bf, t0, L);
  }

  // 5) root (t = 8190): [c, h]
  write_out<<<2, 256, 0, stream>>>(c_arr, h_f32, out);
}
